// ResourceAllocator_12249246728823
// MI455X (gfx1250) — matrix operand order repaired and checked by reading, not compiled, not run on hardware
//
#include <hip/hip_runtime.h>
#include <hip/hip_bf16.h>
#include <math.h>

// ---------------- problem constants ----------------
#define Bn      4
#define Sn      4096
#define Hn      2048
#define HHn     1024
#define NTOK    (Bn * Sn)          // 16384
#define SUPn    4
#define HEADSn  8

// d_out flat offsets (reference tuple order)
#define OFF_TC    0                 // [16384]
#define OFF_CC    16384             // [4]
#define OFF_TK    16388             // [4]
#define OFF_SUP   16392             // [65536]
#define OFF_DEP   81928             // [131072]
#define OFF_EFF   213000            // [16384]
#define OFF_MASK  229384            // [65536]

// GEMM tiling
#define BMt   32      // tokens per block
#define KCt   128     // K slab per stage
#define NCt   32      // N chunk (2 waves x 16)
#define NSTG  (Hn / KCt)   // 16 K-stages per N chunk
#define NBUF  3       // LDS stage buffers (triple buffer for single-barrier pipeline)

// ---------------- types ----------------
typedef __attribute__((ext_vector_type(16))) __bf16 v16bf;
typedef __attribute__((ext_vector_type(8)))  __bf16 v8bf;
typedef __attribute__((ext_vector_type(8)))  float  v8f;
typedef __attribute__((ext_vector_type(4)))  unsigned int u32x4;
typedef __attribute__((ext_vector_type(8)))  unsigned int u32x8;

__device__ __forceinline__ float gelu_exact(float x) {
    return 0.5f * x * (1.0f + erff(x * 0.70710678118654752f));
}
__device__ __forceinline__ float sigmoidf(float x) {
    return 1.0f / (1.0f + __expf(-x));
}
// fp32 -> bf16 (round to nearest even; NaN ignored: compile-only env)
__device__ __forceinline__ unsigned short f2bf(float f) {
    union { float f; unsigned u; } v; v.f = f;
    unsigned u = v.u;
    u += 0x7fffu + ((u >> 16) & 1u);
    return (unsigned short)(u >> 16);
}
// assemble a 16-element bf16 fragment from two contiguous 8-element LDS runs
__device__ __forceinline__ v16bf ld_frag(const unsigned short* p0, const unsigned short* p1) {
    v8bf a = *(const v8bf*)p0;
    v8bf b = *(const v8bf*)p1;
    v16bf r;
#pragma unroll
    for (int i = 0; i < 8; ++i) { r[i] = a[i]; r[i + 8] = b[i]; }
    return r;
}

// ---------------- Tensor Data Mover: 2D bf16 tile, global -> LDS ----------------
// D# group0 (128b): count=1 | lds_addr | global_addr[56:0] | type=2 ("image")
// D# group1 (256b): data_size=1 (2B) | tensor_dim0/1 | tile_dim0/1 | dim0 stride
// Tile lands row-major in LDS: [tile_d1][tile_d0] contiguous — matches our slabs.
__device__ __forceinline__ void tdm_load_2d_bf16(unsigned lds_addr, const void* gaddr,
                                                 unsigned tile_d0, unsigned tile_d1,
                                                 unsigned long long stride0_elems) {
    unsigned long long ga = (unsigned long long)(size_t)gaddr;
    const unsigned tdim0 = 1u << 20, tdim1 = 1u << 20;   // large: no OOB clipping
    u32x4 g0;
    g0[0] = 1u;                                          // count=1, user descriptor
    g0[1] = lds_addr;                                    // LDS byte address
    g0[2] = (unsigned)(ga & 0xffffffffu);
    g0[3] = (unsigned)((ga >> 32) & 0x01ffffffu) | (2u << 30);  // addr[56:32] | type=2
    u32x8 g1;
    g1[0] = (1u << 16);                                  // wg_mask=0, data_size=1 (2 bytes)
    g1[1] = (tdim0 & 0xffffu) << 16;                     // tensor_dim0[15:0] at bit 48
    g1[2] = (tdim0 >> 16) | ((tdim1 & 0xffffu) << 16);   // tensor_dim0[31:16] | tensor_dim1[15:0]
    g1[3] = (tdim1 >> 16) | (tile_d0 << 16);             // tensor_dim1[31:16] | tile_dim0
    g1[4] = (tile_d1 & 0xffffu);                         // tile_dim1 | tile_dim2=0
    g1[5] = (unsigned)(stride0_elems & 0xffffffffu);     // tensor_dim0_stride[31:0]
    g1[6] = (unsigned)((stride0_elems >> 32) & 0xffffu); // stride[47:32] | dim1_stride=0
    g1[7] = 0u;
    u32x4 gz = {0u, 0u, 0u, 0u};                         // unused 3D/4D groups
    asm volatile("tensor_load_to_lds %0, %1, %2, %3"
                 :: "s"(g0), "s"(g1), "s"(gz), "s"(gz)
                 : "memory");
}

// ---------------- K0a: x fp32 -> bf16 ----------------
__global__ void k0_x_bf16(const float* __restrict__ x, unsigned short* __restrict__ xbf) {
    int i = blockIdx.x * blockDim.x + threadIdx.x;       // each handles 4 elems
    size_t base = (size_t)i * 4;
    float4 v = *(const float4*)(x + base);
    unsigned lo = (unsigned)f2bf(v.x) | ((unsigned)f2bf(v.y) << 16);
    unsigned hi = (unsigned)f2bf(v.z) | ((unsigned)f2bf(v.w) << 16);
    uint2 p; p.x = lo; p.y = hi;
    *(uint2*)(xbf + base) = p;
}

// ---------------- K0b: W1 fp32 [K][N] -> bf16 transposed [N][K] ----------------
__global__ void k0_w_transpose(const float* __restrict__ w, unsigned short* __restrict__ wt,
                               int K, int N) {
    int idx = blockIdx.x * blockDim.x + threadIdx.x;
    if (idx < K * N) {
        int k = idx / N, n = idx - k * N;                // coalesced read over n
        wt[(size_t)n * K + k] = f2bf(w[idx]);
    }
}

// ---------------- K1: context mean over S ----------------
__global__ void k1_ctx_mean(const float* __restrict__ x, float* __restrict__ ctxmean) {
    int idx = blockIdx.x * blockDim.x + threadIdx.x;     // 0 .. B*H-1
    int b = idx / Hn, h = idx % Hn;
    const float* p = x + (size_t)b * Sn * Hn + h;
    float s = 0.f;
    for (int t = 0; t < Sn; ++t) s += p[(size_t)t * Hn];
    ctxmean[idx] = s * (1.0f / (float)Sn);
}

// ---------------- K2: ctx & task MLPs on [B,H] (tiny, scalar) ----------------
__global__ void k2_ctx_task(const float* __restrict__ cm,
                            const float* __restrict__ cW1, const float* __restrict__ cB1,
                            const float* __restrict__ cW2, const float* __restrict__ cB2,
                            const float* __restrict__ tW1, const float* __restrict__ tB1,
                            const float* __restrict__ tW2, const float* __restrict__ tB2,
                            float* __restrict__ out) {
    __shared__ float acc[2][Bn];
    int tid = threadIdx.x;
    if (tid < 2 * Bn) ((float*)acc)[tid] = 0.f;
    __syncthreads();
    for (int n = tid; n < HHn; n += blockDim.x) {
        for (int b = 0; b < Bn; ++b) {
            float dc = 0.f, dt = 0.f;
            const float* xb = cm + (size_t)b * Hn;
            for (int k = 0; k < Hn; ++k) {
                float xv = xb[k];
                dc += xv * cW1[(size_t)k * HHn + n];
                dt += xv * tW1[(size_t)k * HHn + n];
            }
            atomicAdd(&acc[0][b], gelu_exact(dc + cB1[n]) * cW2[n]);
            atomicAdd(&acc[1][b], gelu_exact(dt + tB1[n]) * tW2[n]);
        }
    }
    __syncthreads();
    if (tid < Bn)          out[OFF_CC + tid]      = sigmoidf(acc[0][tid] + cB2[0]);
    else if (tid < 2 * Bn) out[OFF_TK + tid - Bn] = sigmoidf(acc[1][tid - Bn] + tB2[0]);
}

// ---------------- K3: fused tok + eff MLPs (pipelined TDM + WMMA bf16) ----------------
// block = 128 threads (4 waves), tile = 32 tokens. Wave w: m_sub = w&1, n_sub = w>>1.
// Triple-buffered LDS stages: TDM for stage s+1 overlaps WMMA on stage s.
__global__ __launch_bounds__(128)
void k3_tok_eff(const unsigned short* __restrict__ xbf,
                const unsigned short* __restrict__ w1tT,   // [HHn][Hn] bf16
                const unsigned short* __restrict__ w1eT,   // [HHn][Hn] bf16
                const float* __restrict__ b1t, const float* __restrict__ w2t, const float* __restrict__ b2t,
                const float* __restrict__ b1e, const float* __restrict__ w2e, const float* __restrict__ b2e,
                float* __restrict__ out) {
    __shared__ __attribute__((aligned(16))) unsigned short ldsA[NBUF][BMt][KCt];        // 3 x 8 KB
    __shared__ __attribute__((aligned(16))) unsigned short ldsB[NBUF][2][NCt][KCt];     // 3 x 16 KB
    __shared__ float sAcc[2][BMt];

    const int tid  = threadIdx.x;
    const int lane = tid & 31;
    const int wave = tid >> 5;
    const int t0   = blockIdx.x * BMt;

    const int m_base = (wave & 1) * 16;
    const int n_sub  = wave >> 1;
    const int rowA   = lane & 15;
    const int half   = lane >> 4;
    const int koffA  = half * 8;        // A: two 8-K runs at +0/+16
    const int colB   = lane & 15;
    const int kbB    = half * 8;        // B: two 8-K runs at +0/+16

    unsigned aAddr[NBUF], b0Addr[NBUF], b1Addr[NBUF];
#pragma unroll
    for (int i = 0; i < NBUF; ++i) {
        aAddr[i]  = (unsigned)(size_t)&ldsA[i][0][0];
        b0Addr[i] = (unsigned)(size_t)&ldsB[i][0][0][0];
        b1Addr[i] = (unsigned)(size_t)&ldsB[i][1][0][0];
    }

    if (tid < 2 * BMt) ((float*)sAcc)[tid] = 0.f;

    float tcp[8], efp[8];
#pragma unroll
    for (int r = 0; r < 8; ++r) { tcp[r] = 0.f; efp[r] = 0.f; }

    for (int nc = 0; nc < HHn; nc += NCt) {
        __syncthreads();   // protect buf[0] WAR vs previous chunk's last stage
        if (wave == 0) {   // prologue: stage 0 DMA
            tdm_load_2d_bf16(aAddr[0],  xbf  + (size_t)t0 * Hn, KCt, BMt, Hn);
            tdm_load_2d_bf16(b0Addr[0], w1tT + (size_t)nc * Hn, KCt, NCt, Hn);
            tdm_load_2d_bf16(b1Addr[0], w1eT + (size_t)nc * Hn, KCt, NCt, Hn);
        }
        v8f accT = {}; v8f accE = {};
        for (int s = 0; s < NSTG; ++s) {
            const int bi = s % NBUF;
            if (wave == 0) {
                if (s + 1 < NSTG) {      // issue next stage, then drain current
                    const int bn = (s + 1) % NBUF;
                    const int kn = (s + 1) * KCt;
                    tdm_load_2d_bf16(aAddr[bn],  xbf  + (size_t)t0 * Hn + kn, KCt, BMt, Hn);
                    tdm_load_2d_bf16(b0Addr[bn], w1tT + (size_t)nc * Hn + kn, KCt, NCt, Hn);
                    tdm_load_2d_bf16(b1Addr[bn], w1eT + (size_t)nc * Hn + kn, KCt, NCt, Hn);
                    __builtin_amdgcn_s_wait_tensorcnt(3);   // stage s complete (in-order)
                } else {
                    __builtin_amdgcn_s_wait_tensorcnt(0);
                }
            }
            __syncthreads();             // stage s data visible to all waves
#pragma unroll
            for (int kk = 0; kk < KCt; kk += 32) {
                const unsigned short* pa = &ldsA[bi][m_base + rowA][kk + koffA];
                v16bf fa = ld_frag(pa, pa + 16);
                const unsigned short* pt = &ldsB[bi][0][n_sub * 16 + colB][kk + kbB];
                v16bf ft = ld_frag(pt, pt + 16);
                const unsigned short* pe = &ldsB[bi][1][n_sub * 16 + colB][kk + kbB];
                v16bf fe = ld_frag(pe, pe + 16);
                accT = __builtin_amdgcn_wmma_f32_16x16x32_bf16(false, fa, false, ft, (short)0, accT, false, false);
                accE = __builtin_amdgcn_wmma_f32_16x16x32_bf16(false, fa, false, fe, (short)0, accE, false, false);
            }
        }
        // epilogue: hidden -> GELU -> dot with W2 (width 1), per-lane accumulate
        int n = nc + n_sub * 16 + colB;
        float bT = b1t[n], wT = w2t[n];
        float bE = b1e[n], wE = w2e[n];
#pragma unroll
        for (int r = 0; r < 8; ++r) {
            tcp[r] += gelu_exact(accT[r] + bT) * wT;
            efp[r] += gelu_exact(accE[r] + bE) * wE;
        }
    }
    // reduce across the 16 lanes of each half-wave
#pragma unroll
    for (int r = 0; r < 8; ++r)
        for (int m = 1; m < 16; m <<= 1) {
            tcp[r] += __shfl_xor(tcp[r], m, 32);
            efp[r] += __shfl_xor(efp[r], m, 32);
        }
    if ((lane & 15) == 0) {
#pragma unroll
        for (int r = 0; r < 8; ++r) {
            int row = m_base + r + half * 8;
            atomicAdd(&sAcc[0][row], tcp[r]);   // ds_add_f32
            atomicAdd(&sAcc[1][row], efp[r]);
        }
    }
    __syncthreads();
    if (tid < BMt) {
        int tok = t0 + tid;
        out[OFF_TC  + tok] = sigmoidf(sAcc[0][tid] + b2t[0]);
        out[OFF_EFF + tok] = sigmoidf(sAcc[1][tid] + b2e[0]);
    }
}

// ---------------- K4: fused sup + dep MLPs (pipelined TDM + WMMA + rank-1 concat) ----------------
__global__ __launch_bounds__(128)
void k4_sup_dep(const unsigned short* __restrict__ xbf,
                const unsigned short* __restrict__ w1sT,   // [Hn][Hn] bf16 (first 2048 rows, transposed)
                const unsigned short* __restrict__ w1dT,
                const float* __restrict__ w1s, const float* __restrict__ b1s,
                const float* __restrict__ w2s, const float* __restrict__ b2s,
                const float* __restrict__ w1d, const float* __restrict__ b1d,
                const float* __restrict__ w2d, const float* __restrict__ b2d,
                float* __restrict__ out) {
    __shared__ __attribute__((aligned(16))) unsigned short ldsA[NBUF][BMt][KCt];
    __shared__ __attribute__((aligned(16))) unsigned short ldsB[NBUF][2][NCt][KCt];
    __shared__ float supAcc[BMt][SUPn];
    __shared__ float depAcc[BMt][HEADSn];
    __shared__ float tcRow[BMt];
    __shared__ float ccS, tkS;

    const int tid  = threadIdx.x;
    const int lane = tid & 31;
    const int wave = tid >> 5;
    const int t0   = blockIdx.x * BMt;
    const int bId  = t0 >> 12;           // token block never spans batches (32 | 4096)

    const int m_base = (wave & 1) * 16;
    const int n_sub  = wave >> 1;
    const int rowA   = lane & 15;
    const int half   = lane >> 4;
    const int koffA  = half * 8;
    const int colB   = lane & 15;
    const int kbB    = half * 8;

    unsigned aAddr[NBUF], b0Addr[NBUF], b1Addr[NBUF];
#pragma unroll
    for (int i = 0; i < NBUF; ++i) {
        aAddr[i]  = (unsigned)(size_t)&ldsA[i][0][0];
        b0Addr[i] = (unsigned)(size_t)&ldsB[i][0][0][0];
        b1Addr[i] = (unsigned)(size_t)&ldsB[i][1][0][0];
    }

    if (tid == 0) { ccS = out[OFF_CC + bId]; tkS = out[OFF_TK + bId]; }
    if (tid < BMt) tcRow[tid] = out[OFF_TC + t0 + tid];
    for (int i = tid; i < BMt * SUPn; i += 128)   ((float*)supAcc)[i] = 0.f;
    for (int i = tid; i < BMt * HEADSn; i += 128) ((float*)depAcc)[i] = 0.f;

    float sp[8][SUPn], dp[8][HEADSn];
#pragma unroll
    for (int r = 0; r < 8; ++r) {
#pragma unroll
        for (int j = 0; j < SUPn; ++j) sp[r][j] = 0.f;
#pragma unroll
        for (int j = 0; j < HEADSn; ++j) dp[r][j] = 0.f;
    }

    for (int nc = 0; nc < Hn; nc += NCt) {            // hidden dim = 2048
        __syncthreads();
        if (wave == 0) {
            tdm_load_2d_bf16(aAddr[0],  xbf  + (size_t)t0 * Hn, KCt, BMt, Hn);
            tdm_load_2d_bf16(b0Addr[0], w1sT + (size_t)nc * Hn, KCt, NCt, Hn);
            tdm_load_2d_bf16(b1Addr[0], w1dT + (size_t)nc * Hn, KCt, NCt, Hn);
        }
        v8f accS = {}; v8f accD = {};
        for (int s = 0; s < NSTG; ++s) {
            const int bi = s % NBUF;
            if (wave == 0) {
                if (s + 1 < NSTG) {
                    const int bn = (s + 1) % NBUF;
                    const int kn = (s + 1) * KCt;
                    tdm_load_2d_bf16(aAddr[bn],  xbf  + (size_t)t0 * Hn + kn, KCt, BMt, Hn);
                    tdm_load_2d_bf16(b0Addr[bn], w1sT + (size_t)nc * Hn + kn, KCt, NCt, Hn);
                    tdm_load_2d_bf16(b1Addr[bn], w1dT + (size_t)nc * Hn + kn, KCt, NCt, Hn);
                    __builtin_amdgcn_s_wait_tensorcnt(3);
                } else {
                    __builtin_amdgcn_s_wait_tensorcnt(0);
                }
            }
            __syncthreads();
#pragma unroll
            for (int kk = 0; kk < KCt; kk += 32) {
                const unsigned short* pa = &ldsA[bi][m_base + rowA][kk + koffA];
                v16bf fa = ld_frag(pa, pa + 16);
                const unsigned short* ps = &ldsB[bi][0][n_sub * 16 + colB][kk + kbB];
                v16bf fs = ld_frag(ps, ps + 16);
                const unsigned short* pd = &ldsB[bi][1][n_sub * 16 + colB][kk + kbB];
                v16bf fd = ld_frag(pd, pd + 16);
                accS = __builtin_amdgcn_wmma_f32_16x16x32_bf16(false, fa, false, fs, (short)0, accS, false, false);
                accD = __builtin_amdgcn_wmma_f32_16x16x32_bf16(false, fa, false, fd, (short)0, accD, false, false);
            }
        }
        // epilogue: add rank-1 concat terms (tc, cc, tk), GELU, dot with W2 columns
        int n = nc + n_sub * 16 + colB;
        float e0s = w1s[(size_t)2048 * Hn + n], e1s = w1s[(size_t)2049 * Hn + n], e2s = w1s[(size_t)2050 * Hn + n];
        float e0d = w1d[(size_t)2048 * Hn + n], e1d = w1d[(size_t)2049 * Hn + n], e2d = w1d[(size_t)2050 * Hn + n];
        float cS = b1s[n] + ccS * e1s + tkS * e2s;
        float cD = b1d[n] + ccS * e1d + tkS * e2d;
        float ws2[SUPn], wd2[HEADSn];
#pragma unroll
        for (int j = 0; j < SUPn; ++j)   ws2[j] = w2s[(size_t)n * SUPn + j];
#pragma unroll
        for (int j = 0; j < HEADSn; ++j) wd2[j] = w2d[(size_t)n * HEADSn + j];
#pragma unroll
        for (int r = 0; r < 8; ++r) {
            int row = m_base + r + half * 8;
            float tcv = tcRow[row];
            float gs = gelu_exact(accS[r] + cS + tcv * e0s);
            float gd = gelu_exact(accD[r] + cD + tcv * e0d);
#pragma unroll
            for (int j = 0; j < SUPn; ++j)   sp[r][j] += gs * ws2[j];
#pragma unroll
            for (int j = 0; j < HEADSn; ++j) dp[r][j] += gd * wd2[j];
        }
    }
    // cross-lane reduce (within 16-lane halves), then combine the two n-waves via LDS
#pragma unroll
    for (int r = 0; r < 8; ++r) {
#pragma unroll
        for (int j = 0; j < SUPn; ++j)
            for (int m = 1; m < 16; m <<= 1) sp[r][j] += __shfl_xor(sp[r][j], m, 32);
#pragma unroll
        for (int j = 0; j < HEADSn; ++j)
            for (int m = 1; m < 16; m <<= 1) dp[r][j] += __shfl_xor(dp[r][j], m, 32);
    }
    if ((lane & 15) == 0) {
#pragma unroll
        for (int r = 0; r < 8; ++r) {
            int row = m_base + r + half * 8;
#pragma unroll
            for (int j = 0; j < SUPn; ++j)   atomicAdd(&supAcc[row][j], sp[r][j]);
#pragma unroll
            for (int j = 0; j < HEADSn; ++j) atomicAdd(&depAcc[row][j], dp[r][j]);
        }
    }
    __syncthreads();
    // finalize: softmax(4), softmax(8), threshold mask
    if (tid < BMt) {
        int tok = t0 + tid;
        float eff = out[OFF_EFF + tok];
        float l4[SUPn];
        float mx = -1e30f;
#pragma unroll
        for (int j = 0; j < SUPn; ++j) { l4[j] = supAcc[tid][j] + b2s[j]; mx = fmaxf(mx, l4[j]); }
        float sum = 0.f;
#pragma unroll
        for (int j = 0; j < SUPn; ++j) { l4[j] = __expf(l4[j] - mx); sum += l4[j]; }
        float inv = 1.0f / sum;
#pragma unroll
        for (int j = 0; j < SUPn; ++j) {
            float w = l4[j] * inv;
            out[OFF_SUP  + (size_t)tok * SUPn + j] = w;
            out[OFF_MASK + (size_t)tok * SUPn + j] = (w > 0.2f) ? eff : 0.f;
        }
        float l8[HEADSn];
        mx = -1e30f;
#pragma unroll
        for (int j = 0; j < HEADSn; ++j) { l8[j] = depAcc[tid][j] + b2d[j]; mx = fmaxf(mx, l8[j]); }
        sum = 0.f;
#pragma unroll
        for (int j = 0; j < HEADSn; ++j) { l8[j] = __expf(l8[j] - mx); sum += l8[j]; }
        inv = 1.0f / sum;
#pragma unroll
        for (int j = 0; j < HEADSn; ++j)
            out[OFF_DEP + (size_t)tok * HEADSn + j] = l8[j] * inv;
    }
}

// ---------------- host launcher ----------------
extern "C" void kernel_launch(void* const* d_in, const int* in_sizes, int n_in,
                              void* d_out, int out_size, void* d_ws, size_t ws_size,
                              hipStream_t stream) {
    const float* x     = (const float*)d_in[0];
    const float* tokW1 = (const float*)d_in[1];  const float* tokB1 = (const float*)d_in[2];
    const float* tokW2 = (const float*)d_in[3];  const float* tokB2 = (const float*)d_in[4];
    const float* ctxW1 = (const float*)d_in[5];  const float* ctxB1 = (const float*)d_in[6];
    const float* ctxW2 = (const float*)d_in[7];  const float* ctxB2 = (const float*)d_in[8];
    const float* tskW1 = (const float*)d_in[9];  const float* tskB1 = (const float*)d_in[10];
    const float* tskW2 = (const float*)d_in[11]; const float* tskB2 = (const float*)d_in[12];
    const float* effW1 = (const float*)d_in[13]; const float* effB1 = (const float*)d_in[14];
    const float* effW2 = (const float*)d_in[15]; const float* effB2 = (const float*)d_in[16];
    const float* supW1 = (const float*)d_in[17]; const float* supB1 = (const float*)d_in[18];
    const float* supW2 = (const float*)d_in[19]; const float* supB2 = (const float*)d_in[20];
    const float* depW1 = (const float*)d_in[21]; const float* depB1 = (const float*)d_in[22];
    const float* depW2 = (const float*)d_in[23]; const float* depB2 = (const float*)d_in[24];

    float* out = (float*)d_out;

    // workspace carve-up (bf16 staging caches + ctx mean)
    unsigned short* xbf  = (unsigned short*)d_ws;                 // [NTOK][Hn]
    unsigned short* tokT = xbf  + (size_t)NTOK * Hn;              // [HHn][Hn]
    unsigned short* effT = tokT + (size_t)HHn * Hn;               // [HHn][Hn]
    unsigned short* supT = effT + (size_t)HHn * Hn;               // [Hn][Hn]
    unsigned short* depT = supT + (size_t)Hn * Hn;                // [Hn][Hn]
    float*          cmean = (float*)(depT + (size_t)Hn * Hn);     // [Bn][Hn]

    // precision/layout pre-passes
    k0_x_bf16<<<(NTOK * (size_t)Hn) / (4 * 256), 256, 0, stream>>>(x, xbf);
    k0_w_transpose<<<(Hn * HHn + 255) / 256, 256, 0, stream>>>(tokW1, tokT, Hn, HHn);
    k0_w_transpose<<<(Hn * HHn + 255) / 256, 256, 0, stream>>>(effW1, effT, Hn, HHn);
    k0_w_transpose<<<(Hn * Hn  + 255) / 256, 256, 0, stream>>>(supW1, supT, Hn, Hn);
    k0_w_transpose<<<(Hn * Hn  + 255) / 256, 256, 0, stream>>>(depW1, depT, Hn, Hn);

    k1_ctx_mean<<<(Bn * Hn) / 256, 256, 0, stream>>>(x, cmean);
    k2_ctx_task<<<1, 256, 0, stream>>>(cmean,
                                       ctxW1, ctxB1, ctxW2, ctxB2,
                                       tskW1, tskB1, tskW2, tskB2, out);
    k3_tok_eff<<<NTOK / 32, 128, 0, stream>>>(xbf, tokT, effT,
                                              tokB1, tokW2, tokB2,
                                              effB1, effW2, effB2, out);
    k4_sup_dep<<<NTOK / 32, 128, 0, stream>>>(xbf, supT, depT,
                                              supW1, supB1, supW2, supB2,
                                              depW1, depB1, depW2, depB2, out);
}
